// PointTransformerWithHeads_42477226557777
// MI455X (gfx1250) — matrix operand order repaired and checked by reading, not compiled, not run on hardware
//
#include <hip/hip_runtime.h>
#include <hip/hip_bf16.h>
#include <stdint.h>
#include <cmath>

// ---------------------------------------------------------------------------
// Types
// ---------------------------------------------------------------------------
typedef _Float16 h16;
typedef __attribute__((ext_vector_type(16))) _Float16 v16h;
typedef __attribute__((ext_vector_type(8)))  _Float16 v8h;
typedef __attribute__((ext_vector_type(8)))  float    v8f;
typedef __attribute__((ext_vector_type(4)))  uint32_t u32x4;
typedef __attribute__((ext_vector_type(8)))  int      i32x8;
typedef __attribute__((ext_vector_type(4)))  int      i32x4;
typedef unsigned long long u64;

#define CAT16(a,b) __builtin_shufflevector(a, b, 0,1,2,3,4,5,6,7,8,9,10,11,12,13,14,15)

#if defined(__has_builtin)
# if __has_builtin(__builtin_amdgcn_tensor_load_to_lds) && __has_builtin(__builtin_amdgcn_s_wait_tensorcnt)
#  define HAVE_TDM 1
# endif
#endif
#ifndef HAVE_TDM
# define HAVE_TDM 0
#endif

__device__ __forceinline__ float gelu_f(float x) {
    return 0.5f * x * (1.0f + erff(x * 0.70710678118654752f));
}

// monotonic float<->u32 encoding for atomic max
__device__ __forceinline__ unsigned fenc(float v) {
    unsigned u = __float_as_uint(v);
    return (u & 0x80000000u) ? ~u : (u | 0x80000000u);
}
__device__ __forceinline__ float fdec(unsigned u) {
    return __uint_as_float((u & 0x80000000u) ? (u & 0x7FFFFFFFu) : ~u);
}

#if HAVE_TDM
__device__ __forceinline__ uint32_t lds_off(const void* p) {
    // flat LDS address: low 32 bits carry the LDS byte offset (ISA 10.2)
    return (uint32_t)(uintptr_t)p;
}
// 2D f16 tile load via Tensor Data Mover. tile: tk x tm elems (tk contiguous),
// tensor extents (td0, td1) from tile origin, row stride in elements.
// Toolchain: clang-23 six-arg form (g0, g1, g2, g3, g4, cpol).
__device__ __forceinline__ void tdm_load_2d(uint32_t ldsOff, const h16* gsrc,
                                            int tk, int tm,
                                            int td0, int td1, int strideElems)
{
    uint64_t ga = (uint64_t)(uintptr_t)gsrc;
    u32x4 g0;
    g0[0] = 1u;                                              // count=1 user D#
    g0[1] = ldsOff;                                          // lds_addr
    g0[2] = (uint32_t)ga;                                    // global_addr[31:0]
    g0[3] = (uint32_t)((ga >> 32) & 0x01FFFFFFull) | (2u << 30); // addr[56:32]|type=2
    uint32_t d0 = (uint32_t)td0, d1 = (uint32_t)td1;
    uint64_t s0 = (uint64_t)(uint32_t)strideElems;
    i32x8 g1;
    g1[0] = (int)(1u << 16);                                 // data_size = 2 bytes
    g1[1] = (int)((d0 & 0xFFFFu) << 16);                     // tensor_dim0 lo
    g1[2] = (int)((d0 >> 16) | ((d1 & 0xFFFFu) << 16));      // td0 hi | td1 lo
    g1[3] = (int)((d1 >> 16) | ((uint32_t)tk << 16));        // td1 hi | tile_dim0
    g1[4] = (int)((uint32_t)tm);                             // tile_dim1 (dim2=0)
    g1[5] = (int)(uint32_t)(s0 & 0xFFFFFFFFull);             // stride0 lo32
    g1[6] = (int)(uint32_t)((s0 >> 32) & 0xFFFFull);         // stride0 hi16
    g1[7] = 0;
    i32x4 z4 = {0, 0, 0, 0};                                 // groups 2/3 unused (2D)
    i32x8 z8 = {0, 0, 0, 0, 0, 0, 0, 0};
    __builtin_amdgcn_tensor_load_to_lds(g0, g1, z4, z4, z8, 0);
}
#endif

// ---------------------------------------------------------------------------
// Structure-building kernels (morton + sort + unique/scan)
// ---------------------------------------------------------------------------
__device__ __forceinline__ u64 part1by2(u64 x) {
    x &= 2097151ull;
    x = (x | x << 32) & 8725724278095871ull;
    x = (x | x << 16) & 8725728556220671ull;
    x = (x | x << 8)  & 1157144660301377551ull;
    x = (x | x << 4)  & 1207822528635744451ull;
    x = (x | x << 2)  & 1317624576693539401ull;
    return x;
}

__global__ void k_meta_init(int* meta) {
    if (threadIdx.x == 0) {
        for (int i = 0; i < 16; ++i) meta[i] = 0;
        meta[0] = 32768;   // n at stage 0
        meta[8] = 32768;   // padded L at stage 0
    }
}

__global__ void k_morton(const int* __restrict__ gc, u64* __restrict__ keys, int n) {
    int i = blockIdx.x * 256 + threadIdx.x;
    if (i >= n) return;
    u64 cx = part1by2((u64)(unsigned)gc[i * 3 + 0]);
    u64 cy = part1by2((u64)(unsigned)gc[i * 3 + 1]);
    u64 cz = part1by2((u64)(unsigned)gc[i * 3 + 2]);
    u64 code = cx | (cy << 1) | (cz << 2);
    keys[i] = (code << 17) | (u64)(unsigned)i;   // stable: index in low bits
}

__global__ void k_bsort(u64* __restrict__ key, int j, int k) {
    int i = blockIdx.x * 256 + threadIdx.x;
    int ixj = i ^ j;
    if (ixj > i) {
        bool up = ((i & k) == 0);
        u64 a = key[i], b = key[ixj];
        bool sw = up ? (a > b) : (a < b);
        if (sw) { key[i] = b; key[ixj] = a; }
    }
}

// single-block scan: unique (code>>3) segments -> clusters + next-level keys
__global__ __launch_bounds__(1024)
void k_stage_build(const u64* __restrict__ keys, int shift, int idxmask,
                   int* __restrict__ clusters, u64* __restrict__ nextk,
                   int* __restrict__ meta, int sIdx, int capR)
{
    __shared__ int sh[1024];
    __shared__ int carry_s;
    int t = threadIdx.x;
    if (t == 0) carry_s = 0;
    __syncthreads();
    int n = meta[sIdx];
    for (int base = 0; base < capR; base += 1024) {
        int i = base + t;
        u64 code = 0;
        int flag = 0;
        if (i < n) {
            code = keys[i] >> shift;
            if (i == 0) flag = 1;
            else flag = (((code >> 3) != ((keys[i - 1] >> shift) >> 3)) ? 1 : 0);
        }
        sh[t] = flag;
        __syncthreads();
        for (int off = 1; off < 1024; off <<= 1) {
            int add = (t >= off) ? sh[t - off] : 0;
            __syncthreads();
            sh[t] += add;
            __syncthreads();
        }
        if (i < n) {
            int seg = carry_s + sh[t] - 1;
            int orig = idxmask ? (int)(keys[i] & (u64)(unsigned)idxmask) : i;
            clusters[orig] = seg;
            if (flag) nextk[seg] = code >> 3;
        }
        __syncthreads();
        if (t == 0) carry_s += sh[1023];
        __syncthreads();
    }
    if (t == 0) {
        meta[sIdx + 1] = carry_s;
        meta[8 + sIdx + 1] = ((carry_s + 1023) / 1024) * 1024;
    }
}

__global__ void k_inv_pad(const u64* __restrict__ skeys, int idxmask,
                          int* __restrict__ inv, int* __restrict__ pad,
                          int capL, const int* __restrict__ nptr)
{
    int i = blockIdx.x * 256 + threadIdx.x;
    if (i >= capL) return;
    int n = *nptr;
    int j = (i < n) ? i : (n - 1);
    int o = skeys ? (int)(skeys[j] & (u64)(unsigned)idxmask) : j;
    pad[i] = o;
    if (i < n) inv[o] = i;
}

// ---------------------------------------------------------------------------
// Weight / activation prep
// ---------------------------------------------------------------------------
__global__ void k_castw(const float* __restrict__ W, h16* __restrict__ Wt,
                        int K, int N, int Kp, int Np)
{
    int i = blockIdx.x * 256 + threadIdx.x;
    int tot = Kp * Np;
    if (i >= tot) return;
    int nn = i / Kp, kk = i % Kp;
    float v = (nn < N && kk < K) ? W[(size_t)kk * N + nn] : 0.0f;
    Wt[i] = (h16)v;
}

__global__ void k_pack_feats(const float* __restrict__ f, h16* __restrict__ out, int n) {
    int i = blockIdx.x * 256 + threadIdx.x;
    if (i >= n * 32) return;
    int row = i >> 5, c = i & 31;
    out[i] = (h16)((c < 4) ? f[row * 4 + c] : 0.0f);
}

// ---------------------------------------------------------------------------
// WMMA GEMM: C[M,N] = act(affine(A[M,K] * Bt^T + bias))
// A row-major f16 (lda), Bt = W transposed [Np][Kp] row-major f16.
// Tiles 64x64x32, 4 waves; A/B tiles DMA'd to LDS by wave0 via TDM, double buffered.
// ---------------------------------------------------------------------------
__global__ __launch_bounds__(128)
void k_gemm(const h16* __restrict__ A, int lda,
            const h16* __restrict__ Bt, int ldb,
            const float* __restrict__ bias,
            h16* __restrict__ Cout, float* __restrict__ Cout32, int ldc,
            const int* __restrict__ Mptr, int Kp, int N, int Np,
            const float* __restrict__ affg, const float* __restrict__ affb, float affs,
            int act)
{
    __shared__ __align__(16) h16 sA[2][64 * 32];
    __shared__ __align__(16) h16 sB[2][64 * 32];
    const int M = *Mptr;
    const int m0 = blockIdx.x * 64;
    const int n0 = blockIdx.y * 64;
    if (m0 >= M) return;
    const int lane = threadIdx.x & 31;
    const int wv   = threadIdx.x >> 5;
    const int r16  = lane & 15;
    const int hi   = lane >> 4;
    const int KT   = Kp >> 5;

    v8f acc[4] = {};

#if HAVE_TDM
    const bool issuer = (wv == 0);
    if (issuer) {
        tdm_load_2d(lds_off(&sA[0][0]), A + (size_t)m0 * lda, 32, 64, Kp,      M - m0,  lda);
        tdm_load_2d(lds_off(&sB[0][0]), Bt + (size_t)n0 * ldb, 32, 64, Kp,     Np - n0, ldb);
    }
#endif
    for (int kt = 0; kt < KT; ++kt) {
        const int cur = kt & 1;
#if HAVE_TDM
        if (issuer) {
            if (kt + 1 < KT) {
                const int k1 = (kt + 1) << 5;
                tdm_load_2d(lds_off(&sA[cur ^ 1][0]), A + (size_t)m0 * lda + k1, 32, 64, Kp - k1, M - m0,  lda);
                tdm_load_2d(lds_off(&sB[cur ^ 1][0]), Bt + (size_t)n0 * ldb + k1, 32, 64, Kp - k1, Np - n0, ldb);
                __builtin_amdgcn_s_wait_tensorcnt(2);  // current pair landed
            } else {
                __builtin_amdgcn_s_wait_tensorcnt(0);
            }
        }
        asm volatile("" ::: "memory");
        __syncthreads();
        const h16* pa = &sA[cur][0];
        const h16* pb = &sB[cur][0];
#else
        __syncthreads();
        const int k0 = kt << 5;
        for (int e = threadIdx.x; e < 64 * 32; e += 128) {
            int rr = e >> 5, kk = e & 31;
            int row = m0 + rr;
            sA[0][e] = (row < M) ? A[(size_t)row * lda + k0 + kk] : (h16)0.0f;
            sB[0][e] = Bt[(size_t)(n0 + rr) * ldb + k0 + kk];
        }
        __syncthreads();
        const h16* pa = &sA[0][0];
        const h16* pb = &sB[0][0];
#endif
        // A fragment: 16 rows (this wave) x 32 K, per ISA 16-bit A layout
        const v8h* arow = (const v8h*)(pa + (16 * wv + r16) * 32);
        v16h af = CAT16(arow[hi ? 1 : 0], arow[hi ? 3 : 2]);
#pragma unroll
        for (int t = 0; t < 4; ++t) {
            const v8h* brow = (const v8h*)(pb + (16 * t + r16) * 32);
            v16h bf = CAT16(brow[hi ? 1 : 0], brow[hi ? 3 : 2]);
            acc[t] = __builtin_amdgcn_wmma_f32_16x16x32_f16(
                false, af, false, bf, (short)0, acc[t], false, false);
        }
        __syncthreads();
    }

    // epilogue: bias -> affine(bn) -> activation
#pragma unroll
    for (int t = 0; t < 4; ++t) {
        int col = n0 + 16 * t + r16;
        if (col >= N) continue;
        float bv = bias ? bias[col] : 0.0f;
        float g = 1.0f, b2 = 0.0f;
        if (affg) { g = affg[col]; b2 = affb[col]; }
#pragma unroll
        for (int rr = 0; rr < 8; ++rr) {
            int row = m0 + wv * 16 + rr + hi * 8;
            if (row >= M) continue;
            float v = acc[t][rr] + bv;
            if (affg) v = g * v * affs + b2;
            if (act == 1) v = gelu_f(v);
            else if (act == 2) v = fmaxf(v, 0.0f);
            if (Cout32) Cout32[(size_t)row * ldc + col] = v;
            else        Cout [(size_t)row * ldc + col] = (h16)v;
        }
    }
}

// ---------------------------------------------------------------------------
// Flash attention over one (patch, head, 128-q-row slab). P=1024, hd=16.
// QKV layout [L][3C]; online softmax; WMMA for both QK^T and PV.
// ---------------------------------------------------------------------------
__global__ __launch_bounds__(256)
void k_attn(const h16* __restrict__ QKV, h16* __restrict__ O,
            int C, int H, const int* __restrict__ Lptr)
{
    __shared__ __align__(16) h16 sK[128 * 16];
    __shared__ __align__(16) h16 sVt[16 * 128];
    __shared__ __align__(16) h16 sP[8][16 * 128];
    const int L = *Lptr;
    const int patch = blockIdx.x;
    if ((patch + 1) * 1024 > L) return;
    const int h  = blockIdx.y;
    const int qb = blockIdx.z;
    const int tid = threadIdx.x, lane = tid & 31, wv = tid >> 5;
    const int r16 = lane & 15;
    const int hi  = lane >> 4;
    const size_t base = (size_t)patch * 1024;
    const int ld = 3 * C;

    // Q fragment (scaled by hd^-0.5 = 0.25), K-dim zero-padded 16->32
    const int qrow = qb * 128 + wv * 16 + r16;
    v8h qv = *(const v8h*)(QKV + (base + qrow) * ld + h * 16 + (hi ? 8 : 0));
    v8h qs;
#pragma unroll
    for (int i = 0; i < 8; ++i) qs[i] = (h16)((float)qv[i] * 0.25f);
    v8h z8 = {};
    v16h aq = CAT16(qs, z8);

    float mrun[8], ssum[8];
    v8f oacc = {};
#pragma unroll
    for (int rr = 0; rr < 8; ++rr) { mrun[rr] = -3.0e38f; ssum[rr] = 0.0f; }

    for (int kt = 0; kt < 8; ++kt) {
        {   // cooperative K / V-transposed tile load (128 keys x 16)
            int rrow = tid >> 1, hf = tid & 1;
            const h16* src = QKV + (base + kt * 128 + rrow) * ld + h * 16 + hf * 8;
            *(v8h*)&sK[rrow * 16 + hf * 8] = *(const v8h*)(src + C);
            v8h vv = *(const v8h*)(src + 2 * C);
#pragma unroll
            for (int d = 0; d < 8; ++d) sVt[(hf * 8 + d) * 128 + rrow] = vv[d];
        }
        __syncthreads();

        // S = Q K^T for 8 column tiles of 16 keys
        v8f sfr[8];
#pragma unroll
        for (int t = 0; t < 8; ++t) {
            const v8h* kr = (const v8h*)&sK[(16 * t + r16) * 16];
            v16h bk = CAT16(kr[hi], z8);
            v8f zz = {};
            sfr[t] = __builtin_amdgcn_wmma_f32_16x16x32_f16(
                false, aq, false, bk, (short)0, zz, false, false);
        }

        // online softmax update (rows rr + 8*hi, replicated across 16 lanes)
#pragma unroll
        for (int rr = 0; rr < 8; ++rr) {
            float tmax = sfr[0][rr];
#pragma unroll
            for (int t = 1; t < 8; ++t) tmax = fmaxf(tmax, sfr[t][rr]);
            for (int off = 1; off < 16; off <<= 1)
                tmax = fmaxf(tmax, __shfl_xor(tmax, off, 32));
            float mn = fmaxf(mrun[rr], tmax);
            float scale = __expf(mrun[rr] - mn);
            float psum = 0.0f;
#pragma unroll
            for (int t = 0; t < 8; ++t) {
                float p = __expf(sfr[t][rr] - mn);
                psum += p;
                sP[wv][(rr + (hi ? 8 : 0)) * 128 + 16 * t + r16] = (h16)p;
            }
            for (int off = 1; off < 16; off <<= 1)
                psum += __shfl_xor(psum, off, 32);
            ssum[rr] = ssum[rr] * scale + psum;
            oacc[rr] *= scale;
            mrun[rr] = mn;
        }

        // O += P V   (4 chunks of 32 keys)
#pragma unroll
        for (int c4 = 0; c4 < 4; ++c4) {
            const h16* prow = &sP[wv][r16 * 128];
            v8h p0 = *(const v8h*)(prow + 32 * c4 + (hi ? 8 : 0));
            v8h p1 = *(const v8h*)(prow + 32 * c4 + 16 + (hi ? 8 : 0));
            v16h ap = CAT16(p0, p1);
            const h16* vrow = &sVt[r16 * 128];
            v8h v0 = *(const v8h*)(vrow + 32 * c4 + (hi ? 8 : 0));
            v8h v1 = *(const v8h*)(vrow + 32 * c4 + 16 + (hi ? 8 : 0));
            v16h bv = CAT16(v0, v1);
            oacc = __builtin_amdgcn_wmma_f32_16x16x32_f16(
                false, ap, false, bv, (short)0, oacc, false, false);
        }
        __syncthreads();
    }

#pragma unroll
    for (int rr = 0; rr < 8; ++rr) {
        int row = qb * 128 + wv * 16 + rr + (hi ? 8 : 0);
        float val = oacc[rr] / ssum[rr];
        O[(base + row) * C + h * 16 + r16] = (h16)val;
    }
}

// ---------------------------------------------------------------------------
// Elementwise / reduction helpers
// ---------------------------------------------------------------------------
__global__ __launch_bounds__(256)
void k_ln(const h16* __restrict__ X, h16* __restrict__ Y,
          const float* __restrict__ g, const float* __restrict__ b,
          int C, const int* __restrict__ Nptr)
{
    int n = *Nptr;
    int row = blockIdx.x * 8 + (threadIdx.x >> 5);
    if (row >= n) return;
    int lane = threadIdx.x & 31;
    const h16* xr = X + (size_t)row * C;
    float s = 0.0f, s2 = 0.0f;
    for (int c = lane; c < C; c += 32) { float v = (float)xr[c]; s += v; s2 += v * v; }
    for (int off = 1; off < 32; off <<= 1) { s += __shfl_xor(s, off, 32); s2 += __shfl_xor(s2, off, 32); }
    float mean = s / C;
    float var = s2 / C - mean * mean;
    float isd = rsqrtf(var + 1e-5f);
    h16* yr = Y + (size_t)row * C;
    for (int c = lane; c < C; c += 32)
        yr[c] = (h16)(g[c] * (((float)xr[c] - mean) * isd) + b[c]);
}

__global__ void k_add(h16* __restrict__ X, const h16* __restrict__ Y,
                      int C, const int* __restrict__ Nptr)
{
    long tot = (long)(*Nptr) * C;
    long i = (long)blockIdx.x * 256 + threadIdx.x;
    if (i >= tot) return;
    X[i] = (h16)((float)X[i] + (float)Y[i]);
}

__global__ void k_gather(const h16* __restrict__ X, const int* __restrict__ pad,
                         h16* __restrict__ Y, int C, int rows)
{
    long i = (long)blockIdx.x * 256 + threadIdx.x;
    if (i >= (long)rows * C) return;
    int row = (int)(i / C), c = (int)(i % C);
    Y[i] = X[(size_t)pad[row] * C + c];
}

__global__ void k_scatter_add(h16* __restrict__ X, const h16* __restrict__ Yo,
                              const int* __restrict__ inv, int C,
                              const int* __restrict__ Nptr)
{
    long tot = (long)(*Nptr) * C;
    long i = (long)blockIdx.x * 256 + threadIdx.x;
    if (i >= tot) return;
    int row = (int)(i / C), c = (int)(i % C);
    X[i] = (h16)((float)X[i] + (float)Yo[(size_t)inv[row] * C + c]);
}

__global__ void k_segmax_init(unsigned* __restrict__ seg, int tot) {
    int i = blockIdx.x * 256 + threadIdx.x;
    if (i < tot) seg[i] = 0u;
}
__global__ void k_segmax(const h16* __restrict__ X, const int* __restrict__ cl,
                         unsigned* __restrict__ seg, int C, const int* __restrict__ nPrev)
{
    long tot = (long)(*nPrev) * C;
    long i = (long)blockIdx.x * 256 + threadIdx.x;
    if (i >= tot) return;
    int row = (int)(i / C), c = (int)(i % C);
    atomicMax(&seg[(size_t)cl[row] * C + c], fenc((float)X[i]));
}
__global__ void k_segfin(const unsigned* __restrict__ seg, h16* __restrict__ out,
                         const float* __restrict__ g, const float* __restrict__ b,
                         float s, int C, const int* __restrict__ nCur)
{
    long tot = (long)(*nCur) * C;
    long i = (long)blockIdx.x * 256 + threadIdx.x;
    if (i >= tot) return;
    int c = (int)(i % C);
    float v = fdec(seg[i]);
    v = g[c] * v * s + b[c];
    out[i] = (h16)gelu_f(v);
}

__global__ void k_upcombine(h16* __restrict__ out, const h16* __restrict__ up,
                            const h16* __restrict__ sk, const int* __restrict__ cl,
                            int C, const int* __restrict__ Nptr)
{
    long tot = (long)(*Nptr) * C;
    long i = (long)blockIdx.x * 256 + threadIdx.x;
    if (i >= tot) return;
    int row = (int)(i / C), c = (int)(i % C);
    out[i] = (h16)((float)up[(size_t)cl[row] * C + c] + (float)sk[i]);
}

__global__ void k_concat(const float* __restrict__ sem, const float* __restrict__ off,
                         float* __restrict__ out, int n)
{
    int i = blockIdx.x * 256 + threadIdx.x;
    if (i >= n) return;
    out[i * 5 + 0] = sem[i * 64 + 0];
    out[i * 5 + 1] = sem[i * 64 + 1];
    out[i * 5 + 2] = off[i * 64 + 0];
    out[i * 5 + 3] = off[i * 64 + 1];
    out[i * 5 + 4] = off[i * 64 + 2];
}

// ---------------------------------------------------------------------------
// Host orchestration
// ---------------------------------------------------------------------------
struct PLin { const float* b; h16* wt; int K, N, Kp, Np; };
struct PAff { const float* b; const float* g; };
struct PBlock { PLin cpe1, cpe2, fc1, fc2, proj, qkv; PAff ln1, ln2; };
struct PUp   { PAff bn; PLin proj, skip; PAff skip_bn; };
struct PPool { PAff bn; PLin proj; };
struct PHead { PAff bn; PLin fc1, fc2; };

static const int ENC_CH_[5] = {32, 64, 128, 256, 512};
static const int DEC_CH_[5] = {64, 64, 128, 256, 512};
static const int ENC_D_[5]  = {2, 2, 2, 6, 2};
static const int ENC_H_[5]  = {2, 4, 8, 16, 32};
static const int DEC_D_[4]  = {2, 2, 2, 2};
static const int DEC_H_[4]  = {4, 4, 8, 16};
static const int CAPN_[5]   = {32768, 4096, 512, 64, 8};     // morton bound: 2^(15-3s)
static const int CAPL_[5]   = {32768, 4096, 1024, 1024, 1024};

static inline int rup(int x, int m) { return ((x + m - 1) / m) * m; }

extern "C" void kernel_launch(void* const* d_in, const int* in_sizes, int n_in,
                              void* d_out, int out_size, void* d_ws, size_t ws_size,
                              hipStream_t stream)
{
    (void)in_sizes; (void)n_in; (void)out_size; (void)ws_size;
    const int*   gc    = (const int*)d_in[0];
    const float* feats = (const float*)d_in[1];
    const float BN_S   = 1.0f / sqrtf(1.0f + 1e-3f);
    const float HEAD_S = 1.0f / sqrtf(1.0f + 1e-4f);

    // ---- bump allocator over d_ws (deterministic every call) ----
    char* wsbase = (char*)d_ws;
    size_t wsoff = 0;
    auto alloc = [&](size_t bytes) -> void* {
        void* p = wsbase + wsoff;
        wsoff += (bytes + 255) & ~(size_t)255;
        return p;
    };

    int*  meta   = (int*)alloc(64 * sizeof(int));
    u64*  keysS[5];
    keysS[0] = (u64*)alloc(32768 * 8);
    for (int s = 1; s < 5; ++s) keysS[s] = (u64*)alloc(4096 * 8);
    int *padA[5], *invA[5], *clA[4];
    for (int s = 0; s < 5; ++s) {
        padA[s] = (int*)alloc((size_t)CAPL_[s] * 4);
        invA[s] = (int*)alloc((size_t)CAPN_[s] * 4);
    }
    for (int s = 0; s < 4; ++s) clA[s] = (int*)alloc((size_t)CAPN_[s] * 4);

    h16* feats16 = (h16*)alloc((size_t)32768 * 32 * 2);
    h16* xa  = (h16*)alloc((size_t)32768 * 64 * 2);
    h16* xb  = (h16*)alloc((size_t)32768 * 64 * 2);
    h16* skipB[4];
    for (int s = 0; s < 4; ++s) skipB[s] = (h16*)alloc((size_t)CAPN_[s] * ENC_CH_[s] * 2);
    h16* t1   = (h16*)alloc((size_t)32768 * 64 * 2);
    h16* t2   = (h16*)alloc((size_t)32768 * 64 * 2);
    h16* t3   = (h16*)alloc((size_t)32768 * 64 * 2);
    h16* qkvb = (h16*)alloc((size_t)32768 * 192 * 2);
    h16* hid  = (h16*)alloc((size_t)32768 * 256 * 2);
    unsigned* segbuf = (unsigned*)alloc((size_t)4096 * 64 * 4);
    float* psem = (float*)alloc((size_t)32768 * 64 * 4);
    float* poff = (float*)alloc((size_t)32768 * 64 * 4);

    // ---- param pytree cursor (sorted-key flatten: dec,emb,enc,off,sem) ----
    int cur = 2;
    auto nextf = [&]() -> const float* { return (const float*)d_in[cur++]; };
    auto prep_lin = [&](int K, int N) -> PLin {
        PLin L;
        L.b = nextf();                 // {'b','w'} sorted
        const float* w = nextf();
        L.K = K; L.N = N; L.Kp = rup(K, 32); L.Np = rup(N, 64);
        L.wt = (h16*)alloc((size_t)L.Kp * L.Np * 2);
        int tot = L.Kp * L.Np;
        k_castw<<<(tot + 255) / 256, 256, 0, stream>>>(w, L.wt, K, N, L.Kp, L.Np);
        return L;
    };
    auto prep_aff = [&]() -> PAff { PAff a; a.b = nextf(); a.g = nextf(); return a; };
    auto prep_block = [&](int C) -> PBlock {
        PBlock b;
        b.cpe1 = prep_lin(C, C);     b.cpe2 = prep_lin(C, C);
        b.fc1  = prep_lin(C, 4 * C); b.fc2  = prep_lin(4 * C, C);
        b.ln1  = prep_aff();         b.ln2  = prep_aff();
        b.proj = prep_lin(C, C);     b.qkv  = prep_lin(C, 3 * C);
        return b;
    };

    PBlock decB[4][2]; PUp decU[4];
    for (int s = 0; s < 4; ++s) {
        for (int i = 0; i < DEC_D_[s]; ++i) decB[s][i] = prep_block(DEC_CH_[s]);
        decU[s].bn = prep_aff();
        decU[s].proj = prep_lin(DEC_CH_[s + 1], DEC_CH_[s]);
        decU[s].skip = prep_lin(ENC_CH_[s], DEC_CH_[s]);
        decU[s].skip_bn = prep_aff();
    }
    PAff embBn = prep_aff();
    PLin embLin = prep_lin(4, 32);
    PBlock encB[5][6]; PPool encP[5];
    for (int s = 0; s < 5; ++s) {
        for (int i = 0; i < ENC_D_[s]; ++i) encB[s][i] = prep_block(ENC_CH_[s]);
        if (s > 0) { encP[s].bn = prep_aff(); encP[s].proj = prep_lin(ENC_CH_[s - 1], ENC_CH_[s]); }
    }
    PHead offH; offH.bn = prep_aff(); offH.fc1 = prep_lin(64, 64); offH.fc2 = prep_lin(64, 3);
    PHead semH; semH.bn = prep_aff(); semH.fc1 = prep_lin(64, 64); semH.fc2 = prep_lin(64, 2);

    // ---- structure building ----
    k_meta_init<<<1, 32, 0, stream>>>(meta);
    k_morton<<<128, 256, 0, stream>>>(gc, keysS[0], 32768);
    for (int k = 2; k <= 32768; k <<= 1)
        for (int j = k >> 1; j >= 1; j >>= 1)
            k_bsort<<<128, 256, 0, stream>>>(keysS[0], j, k);
    k_stage_build<<<1, 1024, 0, stream>>>(keysS[0], 17, 0x1FFFF, clA[0], keysS[1], meta, 0, 32768);
    k_inv_pad<<<(CAPL_[0] + 255) / 256, 256, 0, stream>>>(keysS[0], 0x1FFFF, invA[0], padA[0], CAPL_[0], meta + 0);
    for (int s = 1; s < 4; ++s) {
        int capR = rup(CAPN_[s], 1024);
        k_stage_build<<<1, 1024, 0, stream>>>(keysS[s], 0, 0, clA[s], keysS[s + 1], meta, s, capR);
        k_inv_pad<<<(CAPL_[s] + 255) / 256, 256, 0, stream>>>(nullptr, 0, invA[s], padA[s], CAPL_[s], meta + s);
    }
    k_inv_pad<<<(CAPL_[4] + 255) / 256, 256, 0, stream>>>(nullptr, 0, invA[4], padA[4], CAPL_[4], meta + 4);

    // ---- generic launch helpers ----
    auto gemm = [&](const h16* A, int lda, const PLin& L, const int* Mptr, int Mcap,
                    h16* out16, float* out32, int ldc,
                    const float* ag, const float* ab, float as, int act) {
        dim3 g((Mcap + 63) / 64, L.Np / 64);
        k_gemm<<<g, 128, 0, stream>>>(A, lda, L.wt, L.Kp, L.b, out16, out32, ldc,
                                      Mptr, L.Kp, L.N, L.Np, ag, ab, as, act);
    };
    auto addk = [&](h16* X, const h16* Y, int C, const int* Nptr, int capN) {
        long tot = (long)capN * C;
        k_add<<<(unsigned)((tot + 255) / 256), 256, 0, stream>>>(X, Y, C, Nptr);
    };
    auto run_block = [&](const PBlock& B, h16* x, int C, int H, int s) {
        const int* mn = meta + s;
        const int* mL = meta + 8 + s;
        int capN = CAPN_[s], capL = CAPL_[s];
        // CPE: x += cpe2(cpe1(x))
        gemm(x, C, B.cpe1, mn, capN, t1, nullptr, C, nullptr, nullptr, 1.f, 0);
        gemm(t1, C, B.cpe2, mn, capN, t2, nullptr, C, nullptr, nullptr, 1.f, 0);
        addk(x, t2, C, mn, capN);
        // Attention: x += proj(attn(qkv(gather(ln1(x)))))[inv]
        k_ln<<<(capN + 7) / 8, 256, 0, stream>>>(x, t1, B.ln1.g, B.ln1.b, C, mn);
        {
            long tot = (long)capL * C;
            k_gather<<<(unsigned)((tot + 255) / 256), 256, 0, stream>>>(t1, padA[s], t2, C, capL);
        }
        gemm(t2, C, B.qkv, mL, capL, qkvb, nullptr, 3 * C, nullptr, nullptr, 1.f, 0);
        k_attn<<<dim3(capL / 1024, H, 8), 256, 0, stream>>>(qkvb, t3, C, H, mL);
        gemm(t3, C, B.proj, mL, capL, t1, nullptr, C, nullptr, nullptr, 1.f, 0);
        {
            long tot = (long)capN * C;
            k_scatter_add<<<(unsigned)((tot + 255) / 256), 256, 0, stream>>>(x, t1, invA[s], C, mn);
        }
        // MLP: x += fc2(gelu(fc1(ln2(x))))
        k_ln<<<(capN + 7) / 8, 256, 0, stream>>>(x, t2, B.ln2.g, B.ln2.b, C, mn);
        gemm(t2, C, B.fc1, mn, capN, hid, nullptr, 4 * C, nullptr, nullptr, 1.f, 1);
        gemm(hid, 4 * C, B.fc2, mn, capN, t1, nullptr, C, nullptr, nullptr, 1.f, 0);
        addk(x, t1, C, mn, capN);
    };

    // ---- embedding ----
    k_pack_feats<<<(32768 * 32 + 255) / 256, 256, 0, stream>>>(feats, feats16, 32768);
    h16* x = xa; h16* xalt = xb;
    gemm(feats16, 32, embLin, meta + 0, 32768, x, nullptr, 32, embBn.g, embBn.b, BN_S, 1);

    // ---- encoder ----
    for (int s = 0; s < 5; ++s) {
        int C = ENC_CH_[s];
        if (s > 0) {
            int Cp = ENC_CH_[s - 1];
            gemm(x, Cp, encP[s].proj, meta + (s - 1), CAPN_[s - 1], hid, nullptr, C,
                 nullptr, nullptr, 1.f, 0);
            int segTot = CAPN_[s] * C;
            k_segmax_init<<<(segTot + 255) / 256, 256, 0, stream>>>(segbuf, segTot);
            long tot = (long)CAPN_[s - 1] * C;
            k_segmax<<<(unsigned)((tot + 255) / 256), 256, 0, stream>>>(hid, clA[s - 1], segbuf, C, meta + (s - 1));
            k_segfin<<<(segTot + 255) / 256, 256, 0, stream>>>(segbuf, xalt, encP[s].bn.g, encP[s].bn.b, BN_S, C, meta + s);
            h16* tmp = x; x = xalt; xalt = tmp;
        }
        for (int i = 0; i < ENC_D_[s]; ++i) run_block(encB[s][i], x, C, ENC_H_[s], s);
        if (s < 4)
            (void)hipMemcpyAsync(skipB[s], x, (size_t)CAPN_[s] * C * sizeof(h16),
                                 hipMemcpyDeviceToDevice, stream);
    }

    // ---- decoder ----
    for (int s = 3; s >= 0; --s) {
        int C = DEC_CH_[s];
        gemm(x, DEC_CH_[s + 1], decU[s].proj, meta + (s + 1), CAPN_[s + 1], t3, nullptr, C,
             decU[s].bn.g, decU[s].bn.b, BN_S, 1);
        gemm(skipB[s], ENC_CH_[s], decU[s].skip, meta + s, CAPN_[s], t2, nullptr, C,
             decU[s].skip_bn.g, decU[s].skip_bn.b, BN_S, 1);
        long tot = (long)CAPN_[s] * C;
        k_upcombine<<<(unsigned)((tot + 255) / 256), 256, 0, stream>>>(xalt, t3, t2, clA[s], C, meta + s);
        h16* tmp = x; x = xalt; xalt = tmp;
        for (int i = 0; i < DEC_D_[s]; ++i) run_block(decB[s][i], x, C, DEC_H_[s], s);
    }

    // ---- heads ----
    gemm(x, 64, semH.fc1, meta + 0, 32768, t1, nullptr, 64, semH.bn.g, semH.bn.b, HEAD_S, 2);
    gemm(t1, 64, semH.fc2, meta + 0, 32768, nullptr, psem, 64, nullptr, nullptr, 1.f, 0);
    gemm(x, 64, offH.fc1, meta + 0, 32768, t1, nullptr, 64, offH.bn.g, offH.bn.b, HEAD_S, 2);
    gemm(t1, 64, offH.fc2, meta + 0, 32768, nullptr, poff, 64, nullptr, nullptr, 1.f, 0);
    k_concat<<<(32768 + 255) / 256, 256, 0, stream>>>(psem, poff, (float*)d_out, 32768);
}
